// Seq2SeqDecoder_90675349553918
// MI455X (gfx1250) — matrix operand order repaired and checked by reading, not compiled, not run on hardware
//
#include <hip/hip_runtime.h>
#include <hip/hip_bf16.h>
#include <math.h>

#define LATENT 256
#define HID    1024
#define OUTD   128
#define SEQ    96
#define BATCH  512

typedef __attribute__((ext_vector_type(16))) __bf16 v16bf;
typedef __attribute__((ext_vector_type(8)))  __bf16 v8bf;
typedef __attribute__((ext_vector_type(8)))  float  v8f;

union Frag { v16bf v; v8bf h[2]; };

// ---------------- f32 -> bf16 conversion ----------------
__global__ __launch_bounds__(256) void f2bf_kernel(const float* __restrict__ in,
                                                   __bf16* __restrict__ out, int n) {
  int i = blockIdx.x * 256 + threadIdx.x;
  if (i < n) out[i] = (__bf16)in[i];
}

// ---------------- WMMA GEMM ----------------
// C[M,N] = A1[M,K1] @ B1[N,K1]^T (+ A2[M,K2] @ B2[N,K2]^T) + bias[N]
// Row-major A and B (B row n == column n of the mathematical B matrix).
// Waves arranged WAVES_M x WAVES_N; each wave computes an (MI*16) x (NI*16) tile.
// Block tile: (WAVES_M*MI*16) x (WAVES_N*NI*16).
// Optionally also emits a bf16 copy of C (for feeding the next GEMM).
template<int WAVES_M, int WAVES_N, int MI, int NI>
__global__ __launch_bounds__(32 * WAVES_M * WAVES_N) void gemm_bf16_kernel(
    const __bf16* __restrict__ A1, int K1, const __bf16* __restrict__ B1,
    const __bf16* __restrict__ A2, int K2, const __bf16* __restrict__ B2,
    const float* __restrict__ bias,
    float* __restrict__ C, int ldc,
    __bf16* __restrict__ outbf, int ldbf)
{
  const int lane = threadIdx.x & 31;
  const int wave = threadIdx.x >> 5;
  const int wm   = wave % WAVES_M;
  const int wn   = wave / WAVES_M;     // waves in same column (same wn? no: same n-range share B)
  const int r    = lane & 15;
  const int half = lane >> 4;
  const int m0 = blockIdx.x * (WAVES_M * MI * 16) + wm * MI * 16;
  const int n0 = blockIdx.y * (WAVES_N * NI * 16) + wn * NI * 16;

  v8f acc[MI][NI];
  v8f zero = {0.f,0.f,0.f,0.f,0.f,0.f,0.f,0.f};
#pragma unroll
  for (int i = 0; i < MI; ++i)
#pragma unroll
    for (int j = 0; j < NI; ++j) acc[i][j] = zero;

  auto kloop = [&](const __bf16* __restrict__ A, const __bf16* __restrict__ B, int K) {
    for (int k0 = 0; k0 < K; k0 += 32) {
      Frag a[MI], b[NI];
#pragma unroll
      for (int i = 0; i < MI; ++i) {
        // A 16-bit 16x32 layout: elems 0..7 = K(k0+half*8 ..), elems 8..15 = K(k0+16+half*8 ..)
        const __bf16* p = A + (size_t)(m0 + i * 16 + r) * K + k0 + half * 8;
        a[i].h[0] = *(const v8bf*)(p);
        a[i].h[1] = *(const v8bf*)(p + 16);
      }
#pragma unroll
      for (int j = 0; j < NI; ++j) {
        // B 16-bit 32x16 layout (matches A): elems 0..7 = K(k0+half*8 ..), elems 8..15 = K(k0+16+half*8 ..)
        const __bf16* p = B + (size_t)(n0 + j * 16 + r) * K + k0 + half * 8;
        b[j].h[0] = *(const v8bf*)(p);
        b[j].h[1] = *(const v8bf*)(p + 16);
      }
#pragma unroll
      for (int i = 0; i < MI; ++i)
#pragma unroll
        for (int j = 0; j < NI; ++j)
          acc[i][j] = __builtin_amdgcn_wmma_f32_16x16x32_bf16(
              false, a[i].v, false, b[j].v, (short)0, acc[i][j], false, false);
    }
  };

  kloop(A1, B1, K1);
  if (A2) kloop(A2, B2, K2);

#pragma unroll
  for (int i = 0; i < MI; ++i)
#pragma unroll
    for (int j = 0; j < NI; ++j) {
      const int col = n0 + j * 16 + r;
      const float bv = bias ? bias[col] : 0.0f;
#pragma unroll
      for (int e = 0; e < 8; ++e) {
        // C/D layout: VGPR e -> row m0+i*16+half*8+e, col = n0+j*16+(lane&15)
        const int row = m0 + i * 16 + half * 8 + e;
        const float v = acc[i][j][e] + bv;
        C[(size_t)row * ldc + col] = v;
        if (outbf) outbf[(size_t)row * ldbf + col] = (__bf16)v;
      }
    }
}

// ---------------- elementwise helpers ----------------
__device__ __forceinline__ float sigm(float x) { return 1.0f / (1.0f + expf(-x)); }
__device__ __forceinline__ float gelu_exact(float x) {
  return 0.5f * x * (1.0f + erff(x * 0.70710678118654752f));
}

// LayerNorm(row of 2*HID) -> GELU -> split into h (bf16, duplicated to h1,h2) and c (f32, to c1,c2)
__global__ __launch_bounds__(256) void ln_gelu_split_kernel(
    const float* __restrict__ in,
    const float* __restrict__ gamma, const float* __restrict__ beta,
    __bf16* __restrict__ h1, __bf16* __restrict__ h2,
    float* __restrict__ c1, float* __restrict__ c2)
{
  const int row = blockIdx.x;
  const int R = 2 * HID;
  const float* x = in + (size_t)row * R;
  __shared__ float ssum[256], ssq[256];
  float s = 0.f, q = 0.f;
  for (int j = threadIdx.x; j < R; j += 256) { float v = x[j]; s += v; q += v * v; }
  ssum[threadIdx.x] = s; ssq[threadIdx.x] = q;
  __syncthreads();
  for (int off = 128; off > 0; off >>= 1) {
    if ((int)threadIdx.x < off) {
      ssum[threadIdx.x] += ssum[threadIdx.x + off];
      ssq[threadIdx.x]  += ssq[threadIdx.x + off];
    }
    __syncthreads();
  }
  const float mu  = ssum[0] / R;
  const float var = ssq[0] / R - mu * mu;
  const float inv = rsqrtf(var + 1e-5f);
  for (int j = threadIdx.x; j < R; j += 256) {
    float v = gelu_exact((x[j] - mu) * inv * gamma[j] + beta[j]);
    if (j < HID) {
      h1[(size_t)row * HID + j] = (__bf16)v;
      h2[(size_t)row * HID + j] = (__bf16)v;
    } else {
      c1[(size_t)row * HID + (j - HID)] = v;
      c2[(size_t)row * HID + (j - HID)] = v;
    }
  }
}

// LayerNorm(row of HID) -> GELU -> bf16 out
__global__ __launch_bounds__(256) void ln_gelu_kernel(
    const float* __restrict__ in,
    const float* __restrict__ gamma, const float* __restrict__ beta,
    __bf16* __restrict__ out)
{
  const int row = blockIdx.x;
  const int R = HID;
  const float* x = in + (size_t)row * R;
  __shared__ float ssum[256], ssq[256];
  float s = 0.f, q = 0.f;
  for (int j = threadIdx.x; j < R; j += 256) { float v = x[j]; s += v; q += v * v; }
  ssum[threadIdx.x] = s; ssq[threadIdx.x] = q;
  __syncthreads();
  for (int off = 128; off > 0; off >>= 1) {
    if ((int)threadIdx.x < off) {
      ssum[threadIdx.x] += ssum[threadIdx.x + off];
      ssq[threadIdx.x]  += ssq[threadIdx.x + off];
    }
    __syncthreads();
  }
  const float mu  = ssum[0] / R;
  const float var = ssq[0] / R - mu * mu;
  const float inv = rsqrtf(var + 1e-5f);
  for (int j = threadIdx.x; j < R; j += 256) {
    float v = gelu_exact((x[j] - mu) * inv * gamma[j] + beta[j]);
    out[(size_t)row * R + j] = (__bf16)v;
  }
}

// LSTM cell elementwise: gates[512,4096] (i,f,g,o), c in/out f32, h out bf16
__global__ __launch_bounds__(256) void lstm_elem_kernel(
    const float* __restrict__ gates, float* __restrict__ c, __bf16* __restrict__ h)
{
  int idx = blockIdx.x * 256 + threadIdx.x;
  if (idx >= BATCH * HID) return;
  int m = idx >> 10;          // HID == 1024
  int j = idx & (HID - 1);
  const float* g = gates + (size_t)m * 4 * HID;
  float i_ = g[j], f_ = g[HID + j], gg = g[2 * HID + j], o_ = g[3 * HID + j];
  float cn = sigm(f_) * c[idx] + sigm(i_) * tanhf(gg);
  c[idx] = cn;
  h[idx] = (__bf16)(sigm(o_) * tanhf(cn));
}

// ---------------- host-side launch ----------------
extern "C" void kernel_launch(void* const* d_in, const int* in_sizes, int n_in,
                              void* d_out, int out_size, void* d_ws, size_t ws_size,
                              hipStream_t stream) {
  const float* z      = (const float*)d_in[0];
  const float* lin1_w = (const float*)d_in[1];
  const float* lin1_b = (const float*)d_in[2];
  const float* ln1_w  = (const float*)d_in[3];
  const float* ln1_b  = (const float*)d_in[4];
  const float* w_ih0  = (const float*)d_in[5];
  const float* w_hh0  = (const float*)d_in[6];
  const float* b0     = (const float*)d_in[7];
  const float* w_ih1  = (const float*)d_in[8];
  const float* w_hh1  = (const float*)d_in[9];
  const float* b1     = (const float*)d_in[10];
  const float* p1_w   = (const float*)d_in[11];
  const float* p1_b   = (const float*)d_in[12];
  const float* pln_w  = (const float*)d_in[13];
  const float* pln_b  = (const float*)d_in[14];
  const float* p2_w   = (const float*)d_in[15];
  const float* p2_b   = (const float*)d_in[16];
  float* out = (float*)d_out;

  char* ws = (char*)d_ws;
  size_t off = 0;
  auto alloc = [&](size_t bytes) -> void* {
    void* p = ws + off;
    off += (bytes + 255) & ~(size_t)255;
    return p;
  };

  __bf16* lin1_bf  = (__bf16*)alloc((size_t)2 * HID * LATENT * 2);
  __bf16* wih0_bf  = (__bf16*)alloc((size_t)4 * HID * OUTD * 2);
  __bf16* whh0_bf  = (__bf16*)alloc((size_t)4 * HID * HID * 2);
  __bf16* wih1_bf  = (__bf16*)alloc((size_t)4 * HID * HID * 2);
  __bf16* whh1_bf  = (__bf16*)alloc((size_t)4 * HID * HID * 2);
  __bf16* p1_bf    = (__bf16*)alloc((size_t)HID * HID * 2);
  __bf16* p2_bf    = (__bf16*)alloc((size_t)OUTD * HID * 2);
  __bf16* z_bf     = (__bf16*)alloc((size_t)BATCH * LATENT * 2);
  float*  gates    = (float*)alloc((size_t)BATCH * 4 * HID * 4);
  float*  pscratch = (float*)alloc((size_t)BATCH * HID * 4);
  __bf16* h1       = (__bf16*)alloc((size_t)BATCH * HID * 2);
  __bf16* h2       = (__bf16*)alloc((size_t)BATCH * HID * 2);
  __bf16* u        = (__bf16*)alloc((size_t)BATCH * HID * 2);
  __bf16* xbuf     = (__bf16*)alloc((size_t)BATCH * OUTD * 2);
  float*  c1       = (float*)alloc((size_t)BATCH * HID * 4);
  float*  c2       = (float*)alloc((size_t)BATCH * HID * 4);

  auto cvt = [&](const float* src, __bf16* dst, int n) {
    f2bf_kernel<<<(n + 255) / 256, 256, 0, stream>>>(src, dst, n);
  };
  cvt(lin1_w, lin1_bf, 2 * HID * LATENT);
  cvt(w_ih0,  wih0_bf, 4 * HID * OUTD);
  cvt(w_hh0,  whh0_bf, 4 * HID * HID);
  cvt(w_ih1,  wih1_bf, 4 * HID * HID);
  cvt(w_hh1,  whh1_bf, 4 * HID * HID);
  cvt(p1_w,   p1_bf,   HID * HID);
  cvt(p2_w,   p2_bf,   OUTD * HID);
  cvt(z,      z_bf,    BATCH * LATENT);

  // x0 = 0 (bf16 zero is all-zero bits)
  hipMemsetAsync(xbuf, 0, (size_t)BATCH * OUTD * 2, stream);

  // Big tile: 8 waves as 4(M)x2(N), wave tile 32x64 -> block 128x128.
  // 12 b128 loads per 8 WMMAs; 4 waves share each B fragment via L0.
  auto gemm_big = [&](const __bf16* A1, int K1, const __bf16* B1,
                      const __bf16* A2, int K2, const __bf16* B2,
                      const float* bias, float* C, int ldc,
                      __bf16* obf, int ldbf, int M, int N) {
    dim3 grid(M / 128, N / 128);
    gemm_bf16_kernel<4, 2, 2, 4><<<grid, 256, 0, stream>>>(
        A1, K1, B1, A2, K2, B2, bias, C, ldc, obf, ldbf);
  };
  // Small tile: 8 waves as 2(M)x4(N), wave tile 32x32 -> block 64x128
  // (keeps 8 blocks of parallelism on the N=128 p2 GEMM).
  auto gemm_small = [&](const __bf16* A1, int K1, const __bf16* B1,
                        const float* bias, float* C, int ldc,
                        __bf16* obf, int ldbf, int M, int N) {
    dim3 grid(M / 64, N / 128);
    gemm_bf16_kernel<2, 4, 2, 2><<<grid, 256, 0, stream>>>(
        A1, K1, B1, nullptr, 0, nullptr, bias, C, ldc, obf, ldbf);
  };

  // Phase 1: z @ lin1_w^T + lin1_b -> ln -> gelu -> (h, c)
  gemm_big(z_bf, LATENT, lin1_bf, nullptr, 0, nullptr,
           lin1_b, gates /*reuse as [512,2048] scratch*/, 2 * HID,
           nullptr, 0, BATCH, 2 * HID);
  ln_gelu_split_kernel<<<BATCH, 256, 0, stream>>>(gates, ln1_w, ln1_b, h1, h2, c1, c2);

  // Phase 2: 96 sequential steps
  for (int t = 0; t < SEQ; ++t) {
    // LSTM layer 0: gates = x @ w_ih0^T + h1 @ w_hh0^T + b0
    gemm_big(xbuf, OUTD, wih0_bf, h1, HID, whh0_bf, b0,
             gates, 4 * HID, nullptr, 0, BATCH, 4 * HID);
    lstm_elem_kernel<<<(BATCH * HID) / 256, 256, 0, stream>>>(gates, c1, h1);

    // LSTM layer 1: gates = h1 @ w_ih1^T + h2 @ w_hh1^T + b1
    gemm_big(h1, HID, wih1_bf, h2, HID, whh1_bf, b1,
             gates, 4 * HID, nullptr, 0, BATCH, 4 * HID);
    lstm_elem_kernel<<<(BATCH * HID) / 256, 256, 0, stream>>>(gates, c2, h2);

    // p1 -> layernorm -> gelu
    gemm_big(h2, HID, p1_bf, nullptr, 0, nullptr, p1_b,
             pscratch, HID, nullptr, 0, BATCH, HID);
    ln_gelu_kernel<<<BATCH, 256, 0, stream>>>(pscratch, pln_w, pln_b, u);

    // p2: y = u @ p2_w^T + p2_b -> out[:, t, :] (f32) and xbuf (bf16) for next step
    gemm_small(u, HID, p2_bf, p2_b,
               out + (size_t)t * OUTD, SEQ * OUTD, xbuf, OUTD, BATCH, OUTD);
  }
}
